// ScaledDotProduct_45698452029695
// MI455X (gfx1250) — matrix operand order repaired and checked by reading, not compiled, not run on hardware
//
#include <hip/hip_runtime.h>
#include <math.h>

// ---------------------------------------------------------------------------
// Flash-attention forward [B=2,H=12,S=4096,d=64] fp32 -> fp32 on gfx1250.
// Compute-bound (103 GFLOP vs ~100 MB): both matmuls on v_wmma_f32_16x16x32_bf16.
// "S-transposed" formulation: S^T = K*Q^T so each lane owns one query column,
// making the online softmax per-lane scalar (1 shuffle per reduction) and the
// P^T operand of O^T = V^T*P^T buildable in registers (no LDS round trip).
// ---------------------------------------------------------------------------

typedef __bf16 bf16_t;
typedef __attribute__((ext_vector_type(16))) __bf16 v16bf;
typedef __attribute__((ext_vector_type(8)))  __bf16 v8bf;
typedef __attribute__((ext_vector_type(8)))  float  v8f;

#define HEADS 24      // B*H fused
#define SEQ   4096
#define DH    64
#define BM    64      // query rows per block (16 per wave)
#define BN    64      // key rows per iteration
#define NW    4       // waves per block (wave32)

union bf8u  { bf16_t h[8];       v8bf  v; };
union fragu { unsigned int u[8]; v16bf v; };

__device__ __forceinline__ v16bf concat8(v8bf lo, v8bf hi) {
  return __builtin_shufflevector(lo, hi, 0,1,2,3,4,5,6,7,8,9,10,11,12,13,14,15);
}

__device__ __forceinline__ unsigned int pk_bf16(float a, float b) {
  union { bf16_t h[2]; unsigned int u; } x;
  x.h[0] = (bf16_t)a; x.h[1] = (bf16_t)b;
  return x.u;
}

__global__ __launch_bounds__(NW * 32, 1)
void fa_fwd_bf16_wmma(const float* __restrict__ Qg,
                      const float* __restrict__ Kg,
                      const float* __restrict__ Vg,
                      float* __restrict__ Og)
{
  __shared__ __align__(32) bf16_t sK [BN * DH];   // [key][d] row major
  __shared__ __align__(32) bf16_t sVt[DH * BN];   // [d][key] transposed

  const int tid  = threadIdx.x;
  const int lane = tid & 31;
  const int wave = tid >> 5;
  const int q16  = lane & 15;          // this lane's query column
  const int half = lane >> 4;          // key-half ownership within lane pair

  const int hb   = blockIdx.y;         // fused batch*head
  const int qblk = blockIdx.x;
  const size_t headoff = (size_t)hb * SEQ * DH;
  const float scale = 0.125f;          // 1/sqrt(64), folded into Q

  // ---- Q^T B-fragments, loaded once (lanes 0-15: contraction d = kc*32+0..15;
  //      lanes 16-31: d = kc*32+16..31 -> 16 contiguous floats of one Q row) ----
  v16bf qf[2];
  {
    const int qrow = qblk * BM + wave * 16 + q16;
    const float* qp = Qg + headoff + (size_t)qrow * DH;
    #pragma unroll
    for (int kc = 0; kc < 2; ++kc) {
      const float* p = qp + kc * 32 + half * 16;
      float4 f0 = *(const float4*)(p);
      float4 f1 = *(const float4*)(p + 4);
      float4 f2 = *(const float4*)(p + 8);
      float4 f3 = *(const float4*)(p + 12);
      v16bf q;
      q[0]  = (bf16_t)(f0.x * scale); q[1]  = (bf16_t)(f0.y * scale);
      q[2]  = (bf16_t)(f0.z * scale); q[3]  = (bf16_t)(f0.w * scale);
      q[4]  = (bf16_t)(f1.x * scale); q[5]  = (bf16_t)(f1.y * scale);
      q[6]  = (bf16_t)(f1.z * scale); q[7]  = (bf16_t)(f1.w * scale);
      q[8]  = (bf16_t)(f2.x * scale); q[9]  = (bf16_t)(f2.y * scale);
      q[10] = (bf16_t)(f2.z * scale); q[11] = (bf16_t)(f2.w * scale);
      q[12] = (bf16_t)(f3.x * scale); q[13] = (bf16_t)(f3.y * scale);
      q[14] = (bf16_t)(f3.z * scale); q[15] = (bf16_t)(f3.w * scale);
      qf[kc] = q;
    }
  }

  // O^T accumulators: lane = query col, regs = d rows (t*16 + r + 8*half)
  v8f o[4];
  #pragma unroll
  for (int t = 0; t < 4; ++t) o[t] = (v8f){0.f,0.f,0.f,0.f,0.f,0.f,0.f,0.f};
  float m_i = -__builtin_inff();
  float l_i = 0.f;

  for (int kb0 = 0; kb0 < SEQ; kb0 += BN) {
    __syncthreads();   // previous iteration done reading sK/sVt

    // ---- K tile: row-major, half a row per thread, b128 LDS stores ----
    {
      const int row = tid >> 1;
      const int c0  = (tid & 1) * 32;
      const float* kp = Kg + headoff + (size_t)(kb0 + row) * DH + c0;
      if (kb0 + BN < SEQ) __builtin_prefetch(kp + BN * DH, 0, 0);
      bf8u kb[4];
      #pragma unroll
      for (int i = 0; i < 8; ++i) {
        float4 f = *(const float4*)(kp + 4 * i);
        kb[i >> 1].h[(i & 1) * 4 + 0] = (bf16_t)f.x;
        kb[i >> 1].h[(i & 1) * 4 + 1] = (bf16_t)f.y;
        kb[i >> 1].h[(i & 1) * 4 + 2] = (bf16_t)f.z;
        kb[i >> 1].h[(i & 1) * 4 + 3] = (bf16_t)f.w;
      }
      #pragma unroll
      for (int i = 0; i < 4; ++i)
        *(v8bf*)&sK[row * DH + c0 + 8 * i] = kb[i].v;
    }

    // ---- V tile transposed: thread owns 4 d-columns x 8 keys; coalesced
    //      float4 global reads, contiguous b128 stores into sVt[d][key] ----
    {
      const int d4   = (tid & 15) * 4;
      const int key0 = (tid >> 4) * 8;
      const float* vp = Vg + headoff + (size_t)(kb0 + key0) * DH + d4;
      if (kb0 + BN < SEQ) __builtin_prefetch(vp + BN * DH, 0, 0);
      bf8u cb[4];
      #pragma unroll
      for (int k = 0; k < 8; ++k) {
        float4 f = *(const float4*)(vp + (size_t)k * DH);
        cb[0].h[k] = (bf16_t)f.x;
        cb[1].h[k] = (bf16_t)f.y;
        cb[2].h[k] = (bf16_t)f.z;
        cb[3].h[k] = (bf16_t)f.w;
      }
      #pragma unroll
      for (int j = 0; j < 4; ++j)
        *(v8bf*)&sVt[(size_t)(d4 + j) * BN + key0] = cb[j].v;
    }
    __syncthreads();

    // ---- S^T = K * Q^T : 4 key subtiles x 2 k-chunks = 8 WMMAs ----
    // A-fragment of K (contiguous k layout): lane = key row, two b128 loads.
    v8f s[4];
    #pragma unroll
    for (int m = 0; m < 4; ++m) {
      v8f acc = (v8f){0.f,0.f,0.f,0.f,0.f,0.f,0.f,0.f};
      #pragma unroll
      for (int kc = 0; kc < 2; ++kc) {
        const bf16_t* ap = &sK[(size_t)(m * 16 + q16) * DH + kc * 32 + half * 16];
        v16bf afrag = concat8(*(const v8bf*)ap, *(const v8bf*)(ap + 8));
        acc = __builtin_amdgcn_wmma_f32_16x16x32_bf16(
                  false, afrag, false, qf[kc], (short)0, acc, false, false);
      }
      s[m] = acc;   // s[m][r] = score(key = m*16 + r + 8*half, query = q16)
    }

    // ---- online softmax: per-lane scalar, one shuffle per reduction ----
    float mx = -__builtin_inff();
    #pragma unroll
    for (int m = 0; m < 4; ++m)
      #pragma unroll
      for (int r = 0; r < 8; ++r) mx = fmaxf(mx, s[m][r]);
    mx = fmaxf(mx, __shfl_xor(mx, 16));
    const float mnew  = fmaxf(m_i, mx);
    const float alpha = __expf(m_i - mnew);
    float rs = 0.f;
    #pragma unroll
    for (int m = 0; m < 4; ++m) {
      #pragma unroll
      for (int r = 0; r < 8; ++r) {
        const float p = __expf(s[m][r] - mnew);
        s[m][r] = p;
        rs += p;
      }
    }
    rs += __shfl_xor(rs, 16);
    l_i = l_i * alpha + rs;
    m_i = mnew;
    #pragma unroll
    for (int t = 0; t < 4; ++t)
      #pragma unroll
      for (int r = 0; r < 8; ++r) o[t][r] *= alpha;

    // ---- build P^T B-fragments in registers: pack bf16 pairs along keys,
    //      exchange with partner lane (xor 16), select per half ----
    unsigned int pk[4][4], px[4][4];
    #pragma unroll
    for (int m = 0; m < 4; ++m)
      #pragma unroll
      for (int i = 0; i < 4; ++i)
        pk[m][i] = pk_bf16(s[m][2 * i], s[m][2 * i + 1]);
    #pragma unroll
    for (int m = 0; m < 4; ++m)
      #pragma unroll
      for (int i = 0; i < 4; ++i)
        px[m][i] = (unsigned int)__shfl_xor((int)pk[m][i], 16);

    // ---- O^T += V^T * P^T : 2 k-chunks x 4 d-tiles = 8 WMMAs ----
    #pragma unroll
    for (int kc = 0; kc < 2; ++kc) {
      fragu pf;
      #pragma unroll
      for (int i = 0; i < 4; ++i) {
        // elements 0..7  <- keys kc*32 + 16*half + 0..7  (from half-0 lane)
        // elements 8..15 <- keys kc*32 + 16*half + 8..15 (from half-1 lane)
        pf.u[i]     = half ? px[2 * kc + 1][i] : pk[2 * kc][i];
        pf.u[4 + i] = half ? pk[2 * kc + 1][i] : px[2 * kc][i];
      }
      #pragma unroll
      for (int t = 0; t < 4; ++t) {
        const bf16_t* ap = &sVt[(size_t)(t * 16 + q16) * BN + kc * 32 + half * 16];
        v16bf afrag = concat8(*(const v8bf*)ap, *(const v8bf*)(ap + 8));
        o[t] = __builtin_amdgcn_wmma_f32_16x16x32_bf16(
                   false, afrag, false, pf.v, (short)0, o[t], false, false);
      }
    }
  }

  // ---- normalize and store: d-rows are register-contiguous -> float4 stores ----
  const float inv = 1.0f / l_i;
  const int qrow = qblk * BM + wave * 16 + q16;
  float* op = Og + headoff + (size_t)qrow * DH;
  #pragma unroll
  for (int t = 0; t < 4; ++t) {
    float4 lo4 = make_float4(o[t][0] * inv, o[t][1] * inv,
                             o[t][2] * inv, o[t][3] * inv);
    float4 hi4 = make_float4(o[t][4] * inv, o[t][5] * inv,
                             o[t][6] * inv, o[t][7] * inv);
    float* dst = op + t * 16 + half * 8;
    *(float4*)(dst)     = lo4;
    *(float4*)(dst + 4) = hi4;
  }
}

extern "C" void kernel_launch(void* const* d_in, const int* in_sizes, int n_in,
                              void* d_out, int out_size, void* d_ws, size_t ws_size,
                              hipStream_t stream) {
  (void)in_sizes; (void)n_in; (void)out_size; (void)d_ws; (void)ws_size;
  const float* Q = (const float*)d_in[0];
  const float* K = (const float*)d_in[1];
  const float* V = (const float*)d_in[2];
  float* O = (float*)d_out;
  dim3 grid(SEQ / BM, HEADS);
  dim3 block(NW * 32);
  fa_fwd_bf16_wmma<<<grid, block, 0, stream>>>(Q, K, V, O);
}
